// MultiFeatureGCN_66915590472003
// MI455X (gfx1250) — matrix operand order repaired and checked by reading, not compiled, not run on hardware
//
#include <hip/hip_runtime.h>
#include <hip/hip_bf16.h>
#include <cstdint>

// ---------------------------------------------------------------------------
// MultiFeatureGCN for MI455X (gfx1250, wave32, WMMA 16x16x32 bf16)
// ---------------------------------------------------------------------------
#define NNODES 20000
#define NEDGES 320000
#define HID    128
#define HEADS  4

typedef __attribute__((ext_vector_type(16))) __bf16 v16bf;
typedef __attribute__((ext_vector_type(8)))  float  v8f;

union Frag16 { v16bf v; uint4 q[2]; unsigned int u[8]; };

__device__ __forceinline__ unsigned short f2bf(float x) {
  unsigned int u = __float_as_uint(x);
  unsigned int r = (u + 0x7FFFu + ((u >> 16) & 1u)) >> 16;  // RNE
  return (unsigned short)r;
}
__device__ __forceinline__ unsigned int pack2(float a, float b) {
  return (unsigned int)f2bf(a) | ((unsigned int)f2bf(b) << 16);
}

__device__ __forceinline__ float wave_sum(float v) {
#pragma unroll
  for (int o = 16; o > 0; o >>= 1) v += __shfl_xor(v, o, 32);
  return v;
}

// ordered-uint encoding for float atomic max
__device__ __forceinline__ void atomicMaxFloatEnc(float* addr, float val) {
  unsigned int uv  = __float_as_uint(val);
  unsigned int key = (uv & 0x80000000u) ? ~uv : (uv | 0x80000000u);
  atomicMax((unsigned int*)addr, key);
}
__device__ __forceinline__ float decodeMaxEnc(float enc) {
  unsigned int k = __float_as_uint(enc);
  unsigned int u = (k & 0x80000000u) ? (k & 0x7FFFFFFFu) : ~k;
  return __uint_as_float(u);
}

// ---------------------------------------------------------------------------
// WMMA GEMM:  C[M x Nout] = A[M x K] * W[Nout x K]^T  (+bias) (+=C if accum)
// fp32 in memory, bf16 in WMMA, fp32 accumulation.
// block = 256 threads = 8 wave32, tile 128(M) x 128(N), wave tile 32x64.
// BKP=40 (80B rows) -> every 16B sub-tile aligned => ds_{load,store}_b128,
// conflict-free banking (m*20 mod 64 covers all 64 banks across 16 lanes).
// ---------------------------------------------------------------------------
#define BM 128
#define BN 128
#define BK 32
#define BKP 40

// Stage one 128x32 fp32->bf16 tile into LDS. Thread t covers r=t>>1, kk=(t&1)*16.
__device__ __forceinline__ void stage_tile(unsigned short* __restrict__ dstlds,
                                           const float* __restrict__ src, int ld,
                                           int nrows, int row0, int t, int k0, int K,
                                           bool fastK) {
  const int r  = t >> 1;
  const int kk = (t & 1) * 16;
  const int row = row0 + r;
  unsigned short* d = &dstlds[r * BKP + kk];
  if (fastK && row < nrows) {
    const float* a = src + (size_t)row * ld + k0 + kk;
    float4 x0 = *(const float4*)(a + 0);
    float4 x1 = *(const float4*)(a + 4);
    float4 x2 = *(const float4*)(a + 8);
    float4 x3 = *(const float4*)(a + 12);
    uint4 p0, p1;
    p0.x = pack2(x0.x, x0.y); p0.y = pack2(x0.z, x0.w);
    p0.z = pack2(x1.x, x1.y); p0.w = pack2(x1.z, x1.w);
    p1.x = pack2(x2.x, x2.y); p1.y = pack2(x2.z, x2.w);
    p1.z = pack2(x3.x, x3.y); p1.w = pack2(x3.z, x3.w);
    ((uint4*)d)[0] = p0;
    ((uint4*)d)[1] = p1;
  } else if (row < nrows) {
    const float* a = src + (size_t)row * ld + k0 + kk;
#pragma unroll
    for (int j = 0; j < 16; ++j) {
      const int kg = k0 + kk + j;
      d[j] = (kg < K) ? f2bf(a[j]) : (unsigned short)0;
    }
  } else {
    uint4 z = {0u, 0u, 0u, 0u};
    ((uint4*)d)[0] = z;
    ((uint4*)d)[1] = z;
  }
}

__global__ __launch_bounds__(256) void k_gemm_bf16(
    const float* __restrict__ A, int lda,
    const float* __restrict__ W, int ldw,
    const float* __restrict__ bias,
    float* __restrict__ C, int ldc,
    int M, int K, int Nout, int accumulate) {
  __shared__ __align__(16) unsigned short As[BM * BKP];
  __shared__ __align__(16) unsigned short Bs[BN * BKP];

  const int m0   = blockIdx.x * BM;
  const int n0   = blockIdx.y * BN;
  const int t    = threadIdx.x;
  const int lane = t & 31;
  const int wid  = t >> 5;
  const int wm   = (wid >> 1) * 32;  // 0,32,64,96
  const int wn   = (wid & 1) * 64;   // 0,64

  v8f acc[2][4];
#pragma unroll
  for (int i = 0; i < 2; ++i)
#pragma unroll
    for (int j = 0; j < 4; ++j)
#pragma unroll
      for (int r = 0; r < 8; ++r) acc[i][j][r] = 0.0f;

  const bool fastK  = (K & (BK - 1)) == 0;
  const int  ksteps = (K + BK - 1) / BK;
  for (int kt = 0; kt < ksteps; ++kt) {
    const int k0 = kt * BK;
    stage_tile(As, A, lda, M,    m0, t, k0, K, fastK);
    stage_tile(Bs, W, ldw, Nout, n0, t, k0, K, fastK);
    __syncthreads();

    // ---- fragments per ISA 7.12.2 layouts (16b A 16x32 / B 32x16) ----
    const int am  = lane & 15;
    const int akb = (lane < 16) ? 0 : 8;   // A: lanes 16-31 start at K=8
    const int bkb = (lane < 16) ? 0 : 8;   // B: lanes 16-31 start at K=8
    Frag16 af[2], bfr[4];
#pragma unroll
    for (int i = 0; i < 2; ++i) {
      const uint4* p = (const uint4*)&As[(wm + i * 16 + am) * BKP + akb];
      af[i].q[0] = p[0];   // K = akb .. akb+7
      af[i].q[1] = p[2];   // K = 16+akb .. 16+akb+7
    }
#pragma unroll
    for (int j = 0; j < 4; ++j) {
      const uint4* p = (const uint4*)&Bs[(wn + j * 16 + am) * BKP + bkb];
      bfr[j].q[0] = p[0];  // K = bkb .. bkb+7
      bfr[j].q[1] = p[2];  // K = 16+bkb .. 16+bkb+7
    }
#pragma unroll
    for (int i = 0; i < 2; ++i)
#pragma unroll
      for (int j = 0; j < 4; ++j)
        acc[i][j] = __builtin_amdgcn_wmma_f32_16x16x32_bf16(
            false, af[i].v, false, bfr[j].v, (short)0, acc[i][j], false, false);
    __syncthreads();
  }

  // ---- store C (8-VGPR row-striped layout) ----
  const int nl    = lane & 15;
  const int mbase = (lane >> 4) * 8;
#pragma unroll
  for (int i = 0; i < 2; ++i)
#pragma unroll
    for (int j = 0; j < 4; ++j) {
      const int col = n0 + wn + j * 16 + nl;
      if (col >= Nout) continue;
      const float bv = bias ? bias[col] : 0.0f;
#pragma unroll
      for (int r = 0; r < 8; ++r) {
        const int row = m0 + wm + i * 16 + mbase + r;
        if (row < M) {
          const size_t idx = (size_t)row * ldc + col;
          float v = acc[i][j][r] + bv;
          if (accumulate) v += C[idx];
          C[idx] = v;
        }
      }
    }
}

// ---------------------------------------------------------------------------
// Small elementwise / graph kernels
// ---------------------------------------------------------------------------
__global__ void k_fill(float* p, float v, int n) {
  int i = blockIdx.x * 256 + threadIdx.x;
  if (i < n) p[i] = v;
}
__global__ void k_fillu(unsigned int* p, unsigned int v, int n) {
  int i = blockIdx.x * 256 + threadIdx.x;
  if (i < n) p[i] = v;
}
__global__ void k_deg_edges(const int* __restrict__ dst, const float* __restrict__ ew,
                            float* __restrict__ deg, int E) {
  int e = blockIdx.x * 256 + threadIdx.x;
  if (e < E) atomicAdd(&deg[dst[e]], ew[e]);
}
__global__ void k_dinv(float* __restrict__ d, int n) {
  int i = blockIdx.x * 256 + threadIdx.x;
  if (i < n) { float v = d[i]; d[i] = (v > 0.0f) ? rsqrtf(v) : 0.0f; }
}
__global__ void k_sum_ea(const float* __restrict__ ea, float* __restrict__ out, int E) {
  int i = blockIdx.x * 256 + threadIdx.x;
  float v = (i < E) ? ea[i] : 0.0f;
  v = wave_sum(v);
  if ((threadIdx.x & 31) == 0) atomicAdd(out, v);
}
// scal layout: [0]=sum_ea [1]=mean_ea [2..13]=ce[3][4] [14..16]=fw[3]
__global__ void k_prep(const float* __restrict__ lin_edge_W,
                       const float* __restrict__ att_edge,
                       const float* __restrict__ imp, float* scal, int E) {
  int l = threadIdx.x;
  if (l < 12) {
    int i = l >> 2, h = l & 3;
    const float* we = lin_edge_W + i * (HEADS * HID) + h * HID;
    const float* ae = att_edge + (size_t)(i * HEADS + h) * HID;
    float s = 0.0f;
    for (int k = 0; k < HID; ++k) s += we[k] * ae[k];
    scal[2 + l] = s;
  }
  if (l == 0) {
    scal[1] = scal[0] / (float)E;
    float a = imp[0], b = imp[1], c = imp[2];
    float m = fmaxf(a, fmaxf(b, c));
    float ea = expf(a - m), eb = expf(b - m), ec = expf(c - m);
    float s = ea + eb + ec;
    scal[14] = ea / s; scal[15] = eb / s; scal[16] = ec / s;
  }
}
// out = bias[f] + dinv^2 * xw  (self-loop term of GCN)
__global__ void k_gcn_init(float* __restrict__ outp, const float* __restrict__ xw,
                           const float* __restrict__ bias, const float* __restrict__ dinv,
                           int total) {
  int t = blockIdx.x * 256 + threadIdx.x;
  if (t >= total) return;
  int n = t >> 7, f = t & 127;
  float di = dinv[n];
  outp[t] = bias[f] + di * di * xw[t];
}
// wave-per-edge normalized scatter (GCN)
__global__ __launch_bounds__(256) void k_gcn_edge(
    const int* __restrict__ src, const int* __restrict__ dst,
    const float* __restrict__ ew, const float* __restrict__ dinv,
    const float* __restrict__ xw, float* __restrict__ outp, int E) {
  int wid = threadIdx.x >> 5, lane = threadIdx.x & 31;
  int e = blockIdx.x * 8 + wid;
  if (e >= E) return;
  int s = src[e], d = dst[e];
  float nm = dinv[s] * ew[e] * dinv[d];
  int f0 = lane * 4;
  float4 x = *(const float4*)&xw[(size_t)s * HID + f0];
  float* o = &outp[(size_t)d * HID + f0];
  atomicAdd(o + 0, nm * x.x); atomicAdd(o + 1, nm * x.y);
  atomicAdd(o + 2, nm * x.z); atomicAdd(o + 3, nm * x.w);
}
// dst = relu(acc + bias? + resid)
__global__ void k_brr(float* __restrict__ dstp, const float* __restrict__ acc,
                      const float* __restrict__ bias, const float* __restrict__ resid,
                      int total) {
  int t = blockIdx.x * 256 + threadIdx.x;
  if (t >= total) return;
  int f = t & 127;
  float v = acc[t] + (bias ? bias[f] : 0.0f) + resid[t];
  dstp[t] = v > 0.0f ? v : 0.0f;
}
// per-(node,head) attention coefficients  asrc/adst[n*4+h]
__global__ __launch_bounds__(256) void k_attn(
    const float* __restrict__ xw, const float* __restrict__ a_s,
    const float* __restrict__ a_d, float* __restrict__ asrc,
    float* __restrict__ adst, int NH) {
  int wid = threadIdx.x >> 5, lane = threadIdx.x & 31;
  int idx = blockIdx.x * 8 + wid;
  if (idx >= NH) return;
  int n = idx >> 2, h = idx & 3;
  int f0 = lane * 4;
  float4 x  = *(const float4*)&xw[(size_t)n * (HEADS * HID) + h * HID + f0];
  float4 s4 = *(const float4*)&a_s[h * HID + f0];
  float4 d4 = *(const float4*)&a_d[h * HID + f0];
  float ps = x.x * s4.x + x.y * s4.y + x.z * s4.z + x.w * s4.w;
  float pd = x.x * d4.x + x.y * d4.y + x.z * d4.z + x.w * d4.w;
  ps = wave_sum(ps); pd = wave_sum(pd);
  if (lane == 0) { asrc[idx] = ps; adst[idx] = pd; }
}
__global__ void k_gat_logits(const int* __restrict__ src, const int* __restrict__ dst,
                             const float* __restrict__ ea, const float* __restrict__ asrc,
                             const float* __restrict__ adst, const float* __restrict__ scal,
                             int branch, float* __restrict__ e_att,
                             float* __restrict__ maxd, int E, int Nn) {
  int e = blockIdx.x * 256 + threadIdx.x;
  if (e >= E + Nn) return;
  int s, d; float w;
  if (e < E) { s = src[e]; d = dst[e]; w = ea[e]; }
  else       { s = d = e - E; w = scal[1]; }
  const float* ce = scal + 2 + branch * 4;
#pragma unroll
  for (int h = 0; h < 4; ++h) {
    float v = asrc[s * 4 + h] + adst[d * 4 + h] + w * ce[h];
    v = v > 0.0f ? v : 0.2f * v;  // leaky 0.2
    e_att[(size_t)e * 4 + h] = v;
    atomicMaxFloatEnc(&maxd[d * 4 + h], v);
  }
}
__global__ void k_decode_max(float* p, int n) {
  int i = blockIdx.x * 256 + threadIdx.x;
  if (i < n) p[i] = decodeMaxEnc(p[i]);
}
__global__ void k_gat_exp(const int* __restrict__ dst, const float* __restrict__ maxd,
                          float* __restrict__ e_att, float* __restrict__ ssum,
                          int E, int Nn) {
  int e = blockIdx.x * 256 + threadIdx.x;
  if (e >= E + Nn) return;
  int d = (e < E) ? dst[e] : e - E;
#pragma unroll
  for (int h = 0; h < 4; ++h) {
    float v  = e_att[(size_t)e * 4 + h];
    float ex = expf(v - maxd[d * 4 + h]);
    e_att[(size_t)e * 4 + h] = ex;
    atomicAdd(&ssum[d * 4 + h], ex);
  }
}
__global__ __launch_bounds__(256) void k_gat_scatter(
    const int* __restrict__ src, const int* __restrict__ dst,
    const float* __restrict__ e_att, const float* __restrict__ ssum,
    const float* __restrict__ xw, float* __restrict__ outp, int E, int Nn) {
  int wid = threadIdx.x >> 5, lane = threadIdx.x & 31;
  int e = blockIdx.x * 8 + wid;
  if (e >= E + Nn) return;
  int s, d;
  if (e < E) { s = src[e]; d = dst[e]; } else { s = d = e - E; }
  float al[4];
#pragma unroll
  for (int h = 0; h < 4; ++h)
    al[h] = e_att[(size_t)e * 4 + h] / (ssum[d * 4 + h] + 1e-16f);
  int f0 = lane * 4;
  float ax = 0, ay = 0, az = 0, aw = 0;
#pragma unroll
  for (int h = 0; h < 4; ++h) {
    float4 x = *(const float4*)&xw[(size_t)s * (HEADS * HID) + h * HID + f0];
    ax += al[h] * x.x; ay += al[h] * x.y; az += al[h] * x.z; aw += al[h] * x.w;
  }
  float* o = &outp[(size_t)d * HID + f0];
  atomicAdd(o + 0, 0.25f * ax); atomicAdd(o + 1, 0.25f * ay);
  atomicAdd(o + 2, 0.25f * az); atomicAdd(o + 3, 0.25f * aw);
}
// LayerNorm + leaky(0.1) + 0.7/0.3 blend with softmax-weighted embedding; in-place on hc
__global__ __launch_bounds__(256) void k_ln_fuse(
    float* __restrict__ hc, const float* __restrict__ e_all,
    const float* __restrict__ scal, const float* __restrict__ ln_g,
    const float* __restrict__ ln_b, int Nn) {
  int wid = threadIdx.x >> 5, lane = threadIdx.x & 31;
  int n = blockIdx.x * 8 + wid;
  if (n >= Nn) return;
  int f0 = lane * 4;
  float4 x = *(const float4*)&hc[(size_t)n * HID + f0];
  float mu = wave_sum(x.x + x.y + x.z + x.w) * (1.0f / HID);
  float dx = x.x - mu, dy = x.y - mu, dz = x.z - mu, dw = x.w - mu;
  float var = wave_sum(dx * dx + dy * dy + dz * dz + dw * dw) * (1.0f / HID);
  float inv = rsqrtf(var + 1e-5f);
  float fw0 = scal[14], fw1 = scal[15], fw2 = scal[16];
  float4 g = *(const float4*)&ln_g[f0];
  float4 b = *(const float4*)&ln_b[f0];
  const float* er = e_all + (size_t)n * 384;
  float4 o;
  float hv;
  hv = dx * inv * g.x + b.x; hv = hv > 0 ? hv : 0.1f * hv;
  o.x = hv * 0.7f + (fw0 * er[f0] + fw1 * er[128 + f0] + fw2 * er[256 + f0]) * 0.3f;
  hv = dy * inv * g.y + b.y; hv = hv > 0 ? hv : 0.1f * hv;
  o.y = hv * 0.7f + (fw0 * er[f0 + 1] + fw1 * er[129 + f0] + fw2 * er[257 + f0]) * 0.3f;
  hv = dz * inv * g.z + b.z; hv = hv > 0 ? hv : 0.1f * hv;
  o.z = hv * 0.7f + (fw0 * er[f0 + 2] + fw1 * er[130 + f0] + fw2 * er[258 + f0]) * 0.3f;
  hv = dw * inv * g.w + b.w; hv = hv > 0 ? hv : 0.1f * hv;
  o.w = hv * 0.7f + (fw0 * er[f0 + 3] + fw1 * er[131 + f0] + fw2 * er[259 + f0]) * 0.3f;
  *(float4*)&hc[(size_t)n * HID + f0] = o;
}
__global__ void k_final(const float* __restrict__ hf, const float* __restrict__ Wp,
                        const float* __restrict__ bp, float* __restrict__ out, int Nn) {
  int n = blockIdx.x * 256 + threadIdx.x;
  if (n >= Nn) return;
  float a0 = 0, a1 = 0;
  const float* h = hf + (size_t)n * HID;
#pragma unroll 4
  for (int k = 0; k < HID; ++k) { float v = h[k]; a0 += v * Wp[k]; a1 += v * Wp[HID + k]; }
  out[n * 2 + 0] = a0 + bp[0];
  out[n * 2 + 1] = a1 + bp[1];
}

// ---------------------------------------------------------------------------
static void gemm(hipStream_t st, const float* A, int lda, const float* W, int ldw,
                 const float* bias, float* C, int ldc, int M, int K, int Nout, int acc) {
  dim3 g((M + BM - 1) / BM, (Nout + BN - 1) / BN);
  k_gemm_bf16<<<g, 256, 0, st>>>(A, lda, W, ldw, bias, C, ldc, M, K, Nout, acc);
}

extern "C" void kernel_launch(void* const* d_in, const int* in_sizes, int n_in,
                              void* d_out, int out_size, void* d_ws, size_t ws_size,
                              hipStream_t stream) {
  const int N = NNODES, E = NEDGES;
  const int* src = (const int*)d_in[0];
  const int* dst = src + E;
  const float* edge_attr = (const float*)d_in[1];
  const float* feats[3]  = {(const float*)d_in[2], (const float*)d_in[3], (const float*)d_in[4]};
  const int    Ds[3]     = {512, 20, 1024};
  const float* Wt[3]     = {(const float*)d_in[5], (const float*)d_in[7], (const float*)d_in[9]};
  const float* bt[3]     = {(const float*)d_in[6], (const float*)d_in[8], (const float*)d_in[10]};
  const float* gcn1_W = (const float*)d_in[11]; const float* gcn1_b = (const float*)d_in[12];
  const float* gcn2_W = (const float*)d_in[13]; const float* gcn2_b = (const float*)d_in[14];
  const float* gat_W  = (const float*)d_in[15]; const float* gat_b  = (const float*)d_in[16];
  const float* att_s  = (const float*)d_in[17]; const float* att_d  = (const float*)d_in[18];
  const float* lin_eW = (const float*)d_in[19]; const float* att_e  = (const float*)d_in[20];
  const float* cnn_W  = (const float*)d_in[21]; const float* cnn_b  = (const float*)d_in[22];
  const float* imp    = (const float*)d_in[23];
  const float* Wf     = (const float*)d_in[24]; const float* bfv    = (const float*)d_in[25];
  const float* ln_g   = (const float*)d_in[26]; const float* ln_b   = (const float*)d_in[27];
  const float* Wp     = (const float*)d_in[28]; const float* bp     = (const float*)d_in[29];
  float* out = (float*)d_out;

  float* ws = (float*)d_ws;
  size_t off = 0;
  auto alloc = [&](size_t n) { float* p = ws + off; off += n; return p; };
  float* deg   = alloc(N);
  float* dinv  = alloc(N);
  float* scal  = alloc(32);
  float* bufA  = alloc((size_t)N * HID);
  float* bufB  = alloc((size_t)N * HID);
  float* bufC  = alloc((size_t)N * HID);
  float* xw    = alloc((size_t)N * HEADS * HID);
  float* asrc  = alloc((size_t)N * 4);
  float* adst  = alloc((size_t)N * 4);
  float* maxd  = alloc((size_t)N * 4);
  float* ssum  = alloc((size_t)N * 4);
  float* e_att = alloc((size_t)(E + N) * 4);
  float* e_all = alloc((size_t)N * 384);
  float* hc    = alloc((size_t)N * HID);

  const int gN    = (N + 255) / 256;
  const int gNH   = (N * HID + 255) / 256;
  const int gE    = (E + 255) / 256;
  const int gEN   = (E + N + 255) / 256;
  const int gEw   = (E + 7) / 8;
  const int gENw  = (E + N + 7) / 8;
  const int gNrow = (N + 7) / 8;
  const int gN4   = (N * 4 + 255) / 256;
  const int gNHh  = (N * 4 + 7) / 8;

  // ---- degree / normalization / prep scalars ----
  k_fill<<<gN, 256, 0, stream>>>(deg, 1.0f, N);  // self-loop weight
  k_deg_edges<<<gE, 256, 0, stream>>>(dst, edge_attr, deg, E);
  k_dinv<<<gN, 256, 0, stream>>>(deg, N);
  float* dv = deg;  // deg now holds dinv
  (void)dinv;
  k_fill<<<1, 256, 0, stream>>>(scal, 0.0f, 32);
  k_sum_ea<<<gE, 256, 0, stream>>>(edge_attr, &scal[0], E);
  k_prep<<<1, 32, 0, stream>>>(lin_eW, att_e, imp, scal, E);

  // ---- branches ----
  for (int i = 0; i < 3; ++i) {
    // h = feat @ Wt^T + bt            -> bufA
    gemm(stream, feats[i], Ds[i], Wt[i], Ds[i], bt[i], bufA, HID, N, Ds[i], HID, 0);
    // gcn1: xw = h @ W^T
    gemm(stream, bufA, HID, gcn1_W + (size_t)i * HID * HID, HID, nullptr, xw, HID, N, HID, HID, 0);
    k_gcn_init<<<gNH, 256, 0, stream>>>(bufB, xw, gcn1_b + i * HID, dv, N * HID);
    k_gcn_edge<<<gEw, 256, 0, stream>>>(src, dst, edge_attr, dv, xw, bufB, E);
    k_brr<<<gNH, 256, 0, stream>>>(bufB, bufB, nullptr, bufA, N * HID);  // h1 -> bufB
    // gat: xw512 = h1 @ gat_W^T
    gemm(stream, bufB, HID, gat_W + (size_t)i * HEADS * HID * HID, HID, nullptr,
         xw, HEADS * HID, N, HID, HEADS * HID, 0);
    k_attn<<<gNHh, 256, 0, stream>>>(xw, att_s + (size_t)i * HEADS * HID,
                                     att_d + (size_t)i * HEADS * HID, asrc, adst, N * 4);
    k_fillu<<<gN4, 256, 0, stream>>>((unsigned int*)maxd, 0x007FFFFFu, N * 4);  // enc(-inf)
    k_fill<<<gN4, 256, 0, stream>>>(ssum, 0.0f, N * 4);
    k_gat_logits<<<gEN, 256, 0, stream>>>(src, dst, edge_attr, asrc, adst, scal, i,
                                          e_att, maxd, E, N);
    k_decode_max<<<gN4, 256, 0, stream>>>(maxd, N * 4);
    k_gat_exp<<<gEN, 256, 0, stream>>>(dst, maxd, e_att, ssum, E, N);
    k_fill<<<gNH, 256, 0, stream>>>(bufC, 0.0f, N * HID);
    k_gat_scatter<<<gENw, 256, 0, stream>>>(src, dst, e_att, ssum, xw, bufC, E, N);
    k_brr<<<gNH, 256, 0, stream>>>(bufC, bufC, gat_b + i * HID, bufB, N * HID);  // hg -> bufC
    // gcn2: xw = hg @ W^T
    gemm(stream, bufC, HID, gcn2_W + (size_t)i * HID * HID, HID, nullptr, xw, HID, N, HID, HID, 0);
    k_gcn_init<<<gNH, 256, 0, stream>>>(bufA, xw, gcn2_b + i * HID, dv, N * HID);
    k_gcn_edge<<<gEw, 256, 0, stream>>>(src, dst, edge_attr, dv, xw, bufA, E);
    k_brr<<<gNH, 256, 0, stream>>>(bufA, bufA, nullptr, bufC, N * HID);  // h2 -> bufA
    // cnn einsum: e_all[:, i*128:] = h1 @ W0^T + h2 @ W1^T + b
    const float* W0 = cnn_W + (size_t)i * HID * 2 * HID;
    gemm(stream, bufB, HID, W0,        2 * HID, cnn_b + i * HID, e_all + i * HID, 384, N, HID, HID, 0);
    gemm(stream, bufA, HID, W0 + HID,  2 * HID, nullptr,         e_all + i * HID, 384, N, HID, HID, 1);
  }

  // ---- fusion ----
  gemm(stream, e_all, 384, Wf, 384, bfv, hc, HID, N, 384, HID, 0);
  k_ln_fuse<<<gNrow, 256, 0, stream>>>(hc, e_all, scal, ln_g, ln_b, N);
  k_final<<<gN, 256, 0, stream>>>(hc, Wp, bp, out, N);

  (void)in_sizes; (void)n_in; (void)out_size; (void)ws_size;
}
